// MultiHeadAttention_19774029430918
// MI455X (gfx1250) — matrix operand order repaired and checked by reading, not compiled, not run on hardware
//
#include <hip/hip_runtime.h>
#include <cstdint>
#include <cstddef>

// ---------------------------------------------------------------------------
// Types for CDNA5 WMMA (wave32): v_wmma_f32_16x16x32_f16
// ---------------------------------------------------------------------------
typedef _Float16 v16h __attribute__((ext_vector_type(16)));
typedef _Float16 v8h  __attribute__((ext_vector_type(8)));
typedef float    v8f  __attribute__((ext_vector_type(8)));

#define N_HEAD   16
#define DIM      1024
#define HEAD_DIM 64
#define BATCH    4
#define SEQ      2048
#define M_TOTAL  (BATCH * SEQ)   // 8192 rows for the projection GEMMs

// v_exp_f32 is natively base-2; scores carry a folded log2(e) factor so the
// whole online softmax runs in the log2 domain with raw v_exp_f32.
#define LOG2E 1.44269504088896340736f

// ---------------------------------------------------------------------------
// A-fragment builder for V_WMMA_F32_16X16X32_F16.
// 16-bit A matrix 16x32 (MxK), contiguous k order used here (B matches):
//   lane L (0..15): row = L, halves[0..15] = K 0..15
//   lane L (16..31): row = L-16, halves[0..15] = K 16..31
// rowp must point at (row_base + kstep*32) of a K-contiguous f16 row.
// ---------------------------------------------------------------------------
static __device__ inline v16h make_a_frag(const _Float16* rowp, int lane) {
  const int base = (lane >= 16) ? 16 : 0;
  v8h lo = *(const v8h*)(rowp + base);
  v8h hi = *(const v8h*)(rowp + base + 8);
  v16h r;
#pragma unroll
  for (int i = 0; i < 8; ++i) { r[i] = lo[i]; r[8 + i] = hi[i]; }
  return r;
}

static __device__ inline v8f wmma_f16(v16h a, v16h b, v8f c) {
  return __builtin_amdgcn_wmma_f32_16x16x32_f16(false, a, false, b, (short)0, c,
                                                false, false);
}

// ---------------------------------------------------------------------------
// Kernel 1: Y = X @ W^T + b, optional per-head LayerNorm, f16 output.
//   mode 0: Q -> LN(g,beta), fold (1/sqrt(64))*log2(e), store [B,H,S,D]
//   mode 1: K -> LN(g,beta),                             store [B,H,S,D]
//   mode 2: V -> no LN,                                  store [B,H,D,S] (T)
// Block: 256 threads (8 waves). Tile: 128 rows x 64 cols (one head).
// ---------------------------------------------------------------------------
__global__ __launch_bounds__(256)
void mha_proj_ln_kernel(const float* __restrict__ X, const float* __restrict__ W,
                        const float* __restrict__ bias,
                        const float* __restrict__ gamma,
                        const float* __restrict__ beta,
                        _Float16* __restrict__ dstQK,
                        _Float16* __restrict__ dstVt, int mode) {
  __shared__ __align__(32) _Float16 ldsA[128][32];  // 8 KB: X tile (f16)
  __shared__ __align__(32) _Float16 ldsB[64][32];   // 4 KB: W tile (f16)

  const int tid  = threadIdx.x;
  const int lane = tid & 31;
  const int wave = tid >> 5;
  const int hi   = (lane >= 16) ? 1 : 0;
  const int m0   = blockIdx.x * 128;
  const int head = blockIdx.y;
  const int n0   = head * HEAD_DIM;

  v8f acc[4];
#pragma unroll
  for (int nt = 0; nt < 4; ++nt)
#pragma unroll
    for (int i = 0; i < 8; ++i) acc[nt][i] = 0.0f;

  for (int k0 = 0; k0 < DIM; k0 += 32) {
    // ---- stage X tile: 128x32 fp32 -> f16 LDS (16 elems / thread) ----
    {
      const int row = tid >> 1;
      const int seg = (tid & 1) * 16;
      const float* p = X + (size_t)(m0 + row) * DIM + k0 + seg;
      float4 f0 = ((const float4*)p)[0];
      float4 f1 = ((const float4*)p)[1];
      float4 f2 = ((const float4*)p)[2];
      float4 f3 = ((const float4*)p)[3];
      v8h h0 = {(_Float16)f0.x, (_Float16)f0.y, (_Float16)f0.z, (_Float16)f0.w,
                (_Float16)f1.x, (_Float16)f1.y, (_Float16)f1.z, (_Float16)f1.w};
      v8h h1 = {(_Float16)f2.x, (_Float16)f2.y, (_Float16)f2.z, (_Float16)f2.w,
                (_Float16)f3.x, (_Float16)f3.y, (_Float16)f3.z, (_Float16)f3.w};
      *(v8h*)&ldsA[row][seg]     = h0;
      *(v8h*)&ldsA[row][seg + 8] = h1;
    }
    // ---- stage W tile: 64x32 fp32 -> f16 LDS (8 elems / thread) ----
    {
      const int row = tid >> 2;
      const int seg = (tid & 3) * 8;
      const float* p = W + (size_t)(n0 + row) * DIM + k0 + seg;
      float4 f0 = ((const float4*)p)[0];
      float4 f1 = ((const float4*)p)[1];
      v8h h0 = {(_Float16)f0.x, (_Float16)f0.y, (_Float16)f0.z, (_Float16)f0.w,
                (_Float16)f1.x, (_Float16)f1.y, (_Float16)f1.z, (_Float16)f1.w};
      *(v8h*)&ldsB[row][seg] = h0;
    }
    __syncthreads();

    // ---- compute: each wave owns a 16-row strip, full 64 cols ----
    v16h a = make_a_frag(&ldsA[wave * 16 + (lane & 15)][0], lane);
#pragma unroll
    for (int nt = 0; nt < 4; ++nt) {
      // B matrix 32x16 (KxN): lane L holds col L%16, K (L<16? 0..15 : 16..31)
      v16h b = *(const v16h*)&ldsB[nt * 16 + (lane & 15)][hi * 16];
      acc[nt] = wmma_f16(a, b, acc[nt]);
    }
    __syncthreads();
  }

  // ---- epilogue: bias ----
#pragma unroll
  for (int nt = 0; nt < 4; ++nt) {
    const float bv = bias[n0 + nt * 16 + (lane & 15)];
#pragma unroll
    for (int r = 0; r < 8; ++r) acc[nt][r] += bv;
  }

  if (mode <= 1) {
    // per-head LayerNorm over 64 cols; C-layout row (M = r + 8*hi) lives in a
    // fixed VGPR across one 16-lane half -> shuffle masks 1,2,4,8 reduce it.
    float gv[4], btv[4];
#pragma unroll
    for (int nt = 0; nt < 4; ++nt) {
      gv[nt]  = gamma[nt * 16 + (lane & 15)];
      btv[nt] = beta[nt * 16 + (lane & 15)];
    }
    // Q: fold attention scale AND log2(e) so attention uses raw v_exp_f32.
    const float qscale = (mode == 0) ? (0.125f * LOG2E) : 1.0f;
#pragma unroll
    for (int r = 0; r < 8; ++r) {
      float s  = acc[0][r] + acc[1][r] + acc[2][r] + acc[3][r];
      float s2 = acc[0][r] * acc[0][r] + acc[1][r] * acc[1][r] +
                 acc[2][r] * acc[2][r] + acc[3][r] * acc[3][r];
#pragma unroll
      for (int msk = 8; msk >= 1; msk >>= 1) {
        s  += __shfl_xor(s, msk, 32);
        s2 += __shfl_xor(s2, msk, 32);
      }
      const float mu  = s * (1.0f / 64.0f);
      const float var = s2 * (1.0f / 64.0f) - mu * mu;
      const float rs  = rsqrtf(var + 1e-5f);
#pragma unroll
      for (int nt = 0; nt < 4; ++nt)
        acc[nt][r] = ((acc[nt][r] - mu) * rs * gv[nt] + btv[nt]) * qscale;
    }
    // store f16 [B,H,S,D]
#pragma unroll
    for (int r = 0; r < 8; ++r) {
      const int m    = m0 + wave * 16 + r + hi * 8;
      const int b    = m >> 11;
      const int srow = m & (SEQ - 1);
      const size_t base =
          (((size_t)b * N_HEAD + head) * SEQ + srow) * HEAD_DIM;
#pragma unroll
      for (int nt = 0; nt < 4; ++nt)
        dstQK[base + nt * 16 + (lane & 15)] = (_Float16)acc[nt][r];
    }
  } else {
    // V: store transposed f16 [B,H,D,S]; r=0..7 are consecutive s -> v8h pack
    const int s0   = m0 + wave * 16 + hi * 8;
    const int b    = s0 >> 11;
    const int sr0  = s0 & (SEQ - 1);
    const int bh   = b * N_HEAD + head;
#pragma unroll
    for (int nt = 0; nt < 4; ++nt) {
      const int d = nt * 16 + (lane & 15);
      v8h pack;
#pragma unroll
      for (int r = 0; r < 8; ++r) pack[r] = (_Float16)acc[nt][r];
      *(v8h*)&dstVt[((size_t)bh * HEAD_DIM + d) * SEQ + sr0] = pack;
    }
  }
}

// ---------------------------------------------------------------------------
// Kernel 2: flash attention. Grid (SEQ/128, B*H). Each wave: 16 query rows.
// Q [B,H,S,D] f16 (pre-scaled by 0.125*log2e), K [B,H,S,D] f16, V [B,H,D,S].
// Softmax runs in the log2 domain: p = 2^(score - rowmax2).
// ---------------------------------------------------------------------------
__global__ __launch_bounds__(256)
void mha_flash_attn_kernel(const _Float16* __restrict__ Qh,
                           const _Float16* __restrict__ Kh,
                           const _Float16* __restrict__ Vt,
                           float* __restrict__ out) {
  __shared__ __align__(32) _Float16 ldsP[8][16][64];  // 16 KB: P tiles per wave

  const int tid  = threadIdx.x;
  const int lane = tid & 31;
  const int wave = tid >> 5;
  const int hi   = (lane >= 16) ? 1 : 0;
  const int bh   = blockIdx.y;
  const int b    = bh >> 4;
  const int h    = bh & 15;
  const int q0   = blockIdx.x * 128 + wave * 16;

  const _Float16* Qb = Qh + (size_t)bh * SEQ * HEAD_DIM;
  const _Float16* Kb = Kh + (size_t)bh * SEQ * HEAD_DIM;
  const _Float16* Vb = Vt + (size_t)bh * HEAD_DIM * SEQ;

  // Q fragments (D = 64 -> two k-steps), kept in registers for all key chunks
  v16h qa[2];
#pragma unroll
  for (int ks = 0; ks < 2; ++ks)
    qa[ks] =
        make_a_frag(Qb + (size_t)(q0 + (lane & 15)) * HEAD_DIM + ks * 32, lane);

  v8f o[4];
  float mrow[8], lrow[8];
#pragma unroll
  for (int nt = 0; nt < 4; ++nt)
#pragma unroll
    for (int i = 0; i < 8; ++i) o[nt][i] = 0.0f;
#pragma unroll
  for (int r = 0; r < 8; ++r) { mrow[r] = -1e30f; lrow[r] = 0.0f; }

  for (int kc = 0; kc < SEQ; kc += 64) {
    if (kc + 64 < SEQ) {  // hint next chunk into cache (global_prefetch)
      __builtin_prefetch(Kb + (size_t)(kc + 64 + (lane & 15)) * HEAD_DIM, 0, 1);
      __builtin_prefetch(Vb + (size_t)(lane & 15) * SEQ + kc + 64, 0, 1);
    }
    // ---- scores: S = Q x K^T for 64 keys -> 4 accumulators ----
    v8f sc[4];
#pragma unroll
    for (int g = 0; g < 4; ++g) {
#pragma unroll
      for (int i = 0; i < 8; ++i) sc[g][i] = 0.0f;
      const _Float16* kp =
          Kb + (size_t)(kc + g * 16 + (lane & 15)) * HEAD_DIM + hi * 16;
      v16h kb0 = *(const v16h*)kp;          // D 0..31 slice
      v16h kb1 = *(const v16h*)(kp + 32);   // D 32..63 slice
      sc[g] = wmma_f16(qa[0], kb0, sc[g]);
      sc[g] = wmma_f16(qa[1], kb1, sc[g]);
    }

    // ---- online softmax (log2 domain): 16-lane shuffle reductions ----
    float mnew[8], alpha[8], lsum[8];
#pragma unroll
    for (int r = 0; r < 8; ++r) {
      float tm = fmaxf(fmaxf(sc[0][r], sc[1][r]), fmaxf(sc[2][r], sc[3][r]));
#pragma unroll
      for (int msk = 8; msk >= 1; msk >>= 1) tm = fmaxf(tm, __shfl_xor(tm, msk, 32));
      mnew[r]  = fmaxf(mrow[r], tm);
      alpha[r] = __builtin_amdgcn_exp2f(mrow[r] - mnew[r]);  // raw v_exp_f32
      lsum[r]  = 0.0f;
    }
#pragma unroll
    for (int g = 0; g < 4; ++g) {
#pragma unroll
      for (int r = 0; r < 8; ++r) {
        const float p = __builtin_amdgcn_exp2f(sc[g][r] - mnew[r]);
        lsum[r] += p;
        ldsP[wave][r + hi * 8][g * 16 + (lane & 15)] = (_Float16)p;
      }
    }
#pragma unroll
    for (int r = 0; r < 8; ++r) {
      float ls = lsum[r];
#pragma unroll
      for (int msk = 8; msk >= 1; msk >>= 1) ls += __shfl_xor(ls, msk, 32);
      lrow[r] = lrow[r] * alpha[r] + ls;
      mrow[r] = mnew[r];
    }
#pragma unroll
    for (int nt = 0; nt < 4; ++nt)
#pragma unroll
      for (int r = 0; r < 8; ++r) o[nt][r] *= alpha[r];

    // ---- O += P x V (P re-shaped via per-wave LDS tile; DS is in-order) ----
#pragma unroll
    for (int ks = 0; ks < 2; ++ks) {
      v16h pa = make_a_frag(&ldsP[wave][lane & 15][ks * 32], lane);
#pragma unroll
      for (int nt = 0; nt < 4; ++nt) {
        const _Float16* vp = Vb + (size_t)(nt * 16 + (lane & 15)) * SEQ + kc +
                             ks * 32 + hi * 16;
        v16h vb = *(const v16h*)vp;
        o[nt] = wmma_f16(pa, vb, o[nt]);
      }
    }
  }

  // ---- epilogue: normalize, write fp32 [B,S,DIM] ----
#pragma unroll
  for (int r = 0; r < 8; ++r) {
    const float inv  = 1.0f / lrow[r];
    const int   srow = q0 + r + hi * 8;
    float* op = out + ((size_t)b * SEQ + srow) * DIM + h * HEAD_DIM;
#pragma unroll
    for (int nt = 0; nt < 4; ++nt)
      op[nt * 16 + (lane & 15)] = o[nt][r] * inv;
  }
}

// ---------------------------------------------------------------------------
extern "C" void kernel_launch(void* const* d_in, const int* in_sizes, int n_in,
                              void* d_out, int out_size, void* d_ws,
                              size_t ws_size, hipStream_t stream) {
  (void)in_sizes; (void)n_in; (void)out_size; (void)ws_size;
  const float* q   = (const float*)d_in[0];
  const float* k   = (const float*)d_in[1];
  const float* v   = (const float*)d_in[2];
  const float* Wq  = (const float*)d_in[3];
  const float* bq  = (const float*)d_in[4];
  const float* Wk  = (const float*)d_in[5];
  const float* bk  = (const float*)d_in[6];
  const float* Wv  = (const float*)d_in[7];
  const float* bv  = (const float*)d_in[8];
  const float* g1  = (const float*)d_in[9];
  const float* be1 = (const float*)d_in[10];
  const float* g2  = (const float*)d_in[11];
  const float* be2 = (const float*)d_in[12];
  float* out = (float*)d_out;

  const size_t elems = (size_t)BATCH * N_HEAD * SEQ * HEAD_DIM;  // 8.4M
  _Float16* Qh = (_Float16*)d_ws;      // 16 MB  [B,H,S,D], pre-scaled
  _Float16* Kh = Qh + elems;           // 16 MB  [B,H,S,D]
  _Float16* Vt = Kh + elems;           // 16 MB  [B,H,D,S] (transposed)

  dim3 blk(256);
  dim3 gproj(M_TOTAL / 128, N_HEAD, 1);
  mha_proj_ln_kernel<<<gproj, blk, 0, stream>>>(q, Wq, bq, g1, be1, Qh, nullptr, 0);
  mha_proj_ln_kernel<<<gproj, blk, 0, stream>>>(k, Wk, bk, g2, be2, Kh, nullptr, 1);
  mha_proj_ln_kernel<<<gproj, blk, 0, stream>>>(v, Wv, bv, nullptr, nullptr, nullptr, Vt, 2);

  dim3 gattn(SEQ / 128, BATCH * N_HEAD, 1);
  mha_flash_attn_kernel<<<gattn, blk, 0, stream>>>(Qh, Kh, Vt, out);
}
